// SwinTransformerBlock_14448269984377
// MI455X (gfx1250) — matrix operand order repaired and checked by reading, not compiled, not run on hardware
//
#include <hip/hip_runtime.h>
#include <hip/hip_bf16.h>

typedef __attribute__((ext_vector_type(16))) _Float16 v16h;
typedef __attribute__((ext_vector_type(8)))  _Float16 v8h;
typedef __attribute__((ext_vector_type(8)))  float    v8f;
typedef __attribute__((ext_vector_type(4)))  int      v4i;

#define WMMA_F16(a, b, c) \
  __builtin_amdgcn_wmma_f32_16x16x32_f16(false, (a), false, (b), (short)0, (c), false, false)

#define AS1 __attribute__((address_space(1)))
#define AS3 __attribute__((address_space(3)))

#if __has_builtin(__builtin_amdgcn_global_load_async_to_lds_b128)
#define HAVE_ASYNC_LDS 1
#else
#define HAVE_ASYNC_LDS 0
#endif

static constexpr int   kB     = 8;
static constexpr int   kH     = 128;
static constexpr int   kW     = 128;
static constexpr int   kC     = 192;
static constexpr int   kNH    = 6;
static constexpr int   kHD    = 32;          // head dim
static constexpr int   kWS    = 8;
static constexpr int   kSH    = 4;           // shift
static constexpr int   kN     = 64;          // tokens per window
static constexpr int   kWTOT  = kB * 256;    // 2048 windows
static constexpr int   kMTOT  = kWTOT * kN;  // 131072 tokens
static constexpr float kSCALE = 0.17677669529663688f; // 32^-0.5
static constexpr float kEPS   = 1e-5f;

// ---------------------------------------------------------------------------
// Fragment helpers (CDNA5 wave32 WMMA layouts, CDNA5 ISA 7.12.2)
// ---------------------------------------------------------------------------
__device__ __forceinline__ v16h make_v16h(v8h lo, v8h hi) {
  return __builtin_shufflevector(lo, hi, 0, 1, 2, 3, 4, 5, 6, 7,
                                 8, 9, 10, 11, 12, 13, 14, 15);
}

// A-fragment from row-major [M][lda] source (global or LDS)
__device__ __forceinline__ v16h load_frag_a(const _Float16* a, int lda, int lane) {
  const _Float16* p = a + (size_t)(lane & 15) * lda + ((lane >> 4) * 16);
  return make_v16h(*(const v8h*)(p), *(const v8h*)(p + 8));
}

// B-fragment from pre-packed fragment block: lane L's 16 halves contiguous
__device__ __forceinline__ v16h load_frag_b_packed(const _Float16* frag, int lane) {
  const _Float16* p = frag + lane * 16;
  return make_v16h(*(const v8h*)p, *(const v8h*)(p + 8));
}

// B-fragment where B[k][n] is stored with n as the row index (row stride ldb)
__device__ __forceinline__ v16h load_frag_b_rows(const _Float16* bm, int ldb, int lane) {
  const _Float16* p = bm + (size_t)(lane & 15) * ldb + ((lane >> 4) * 16);
  return make_v16h(*(const v8h*)p, *(const v8h*)(p + 8));
}

// async copy of `halves` f16 (multiple of 256) global -> LDS, one wave
__device__ __forceinline__ void async_copy_wave(const _Float16* g, _Float16* l,
                                                int halves, int lane) {
#if HAVE_ASYNC_LDS
  const _Float16* gs = g + lane * 8;   // 16 bytes per lane
  _Float16* ls = l + lane * 8;
  for (int i = 0; i < (halves >> 8); ++i)
    __builtin_amdgcn_global_load_async_to_lds_b128(
        (AS1 v4i*)(gs + i * 256), (AS3 v4i*)(ls + i * 256), 0, 0);
#else
  const v8h* gs = (const v8h*)(g) + lane;
  v8h* ls = (v8h*)(l) + lane;
  for (int i = 0; i < (halves >> 8); ++i) ls[i * 32] = gs[i * 32];
#endif
}

__device__ __forceinline__ void async_wait0() {
#if HAVE_ASYNC_LDS
#if __has_builtin(__builtin_amdgcn_s_wait_asynccnt)
  __builtin_amdgcn_s_wait_asynccnt(0);
#else
  asm volatile("s_wait_asynccnt 0" ::: "memory");
#endif
#endif
}

// ---------------------------------------------------------------------------
// Pack fp32 weight W[K][N] into f16 B-fragment blocks.
// Fragment (nt, kc) at ((nt*(K/32)+kc)*512); (lane,h) =
//   W[kc*32 + 16*(lane>>4) + h][nt*16 + (lane&15)].
// ---------------------------------------------------------------------------
__global__ void pack_b_kernel(const float* __restrict__ src,
                              _Float16* __restrict__ dst, int K, int N) {
  int idx = blockIdx.x * blockDim.x + threadIdx.x;
  const int total = K * N;
  if (idx >= total) return;
  const int h    = idx & 15;
  const int L    = (idx >> 4) & 31;
  const int frag = idx >> 9;
  const int k32  = K >> 5;
  const int nt   = frag / k32;
  const int kc   = frag - nt * k32;
  const int k = kc * 32 + (L >> 4) * 16 + h;
  const int n = nt * 16 + (L & 15);
  dst[idx] = (_Float16)src[(size_t)k * N + n];
}

// ---------------------------------------------------------------------------
// LN1 + cyclic shift(-4,-4) + window partition -> f16 xw[131072][192]
// ---------------------------------------------------------------------------
__global__ void ln1_shift_part_kernel(const float* __restrict__ x,
                                      const float* __restrict__ lw,
                                      const float* __restrict__ lb,
                                      _Float16* __restrict__ xw) {
  const int lane = threadIdx.x & 31;
  const int wv   = threadIdx.x >> 5;
  const int g    = blockIdx.x * (blockDim.x >> 5) + wv;
  const int win  = g >> 6, tok = g & 63;
  const int img  = win >> 8, wi = win & 255;
  const int hs = ((wi >> 4) * kWS + (tok >> 3) + kSH) & (kH - 1);
  const int ws = ((wi & 15) * kWS + (tok & 7) + kSH) & (kW - 1);
  const float* src = x + ((size_t)img * (kH * kW) + hs * kW + ws) * kC;

  float v[6], s = 0.f, ss = 0.f;
#pragma unroll
  for (int i = 0; i < 6; ++i) {
    v[i] = src[lane + 32 * i];
    s += v[i];
    ss += v[i] * v[i];
  }
#pragma unroll
  for (int off = 16; off; off >>= 1) {
    s  += __shfl_xor(s, off, 32);
    ss += __shfl_xor(ss, off, 32);
  }
  const float mu  = s * (1.f / kC);
  const float inv = rsqrtf(ss * (1.f / kC) - mu * mu + kEPS);
  _Float16* dst = xw + (size_t)g * kC;
#pragma unroll
  for (int i = 0; i < 6; ++i) {
    const int c = lane + 32 * i;
    dst[c] = (_Float16)((v[i] - mu) * inv * lw[c] + lb[c]);
  }
}

// ---------------------------------------------------------------------------
// QKV GEMM: [131072,192] x packed[192,576] -> qkv f16
//   Q,K stored [win][which][head][tok][d]; V stored transposed [..][d][tok].
// Wave tile 16x64 (4 accumulators), 24 WMMAs / wave.
// ---------------------------------------------------------------------------
__global__ void qkv_gemm_kernel(const _Float16* __restrict__ xw,
                                const _Float16* __restrict__ wpk,
                                const float* __restrict__ bias,
                                _Float16* __restrict__ qkv) {
  const int lane = threadIdx.x & 31;
  const int wv   = threadIdx.x >> 5;
  const int mt   = blockIdx.x * 4 + wv;
  const int ng   = blockIdx.y;            // 0..8
  const _Float16* A = xw + (size_t)mt * 16 * kC;
  constexpr int K32 = kC / 32;            // 6
  v8f acc[4] = {};
#pragma unroll
  for (int kc = 0; kc < K32; ++kc) {
    v16h a = load_frag_a(A + kc * 32, kC, lane);
    if (kc + 1 < K32)
      __builtin_prefetch(wpk + ((size_t)(ng * 4) * K32 + kc + 1) * 512, 0, 3);
#pragma unroll
    for (int t = 0; t < 4; ++t) {
      v16h b = load_frag_b_packed(wpk + ((size_t)(ng * 4 + t) * K32 + kc) * 512, lane);
      acc[t] = WMMA_F16(a, b, acc[t]);
    }
  }
  const int col = lane & 15;
  const int hi8 = (lane >> 4) * 8;
  const int win = (mt * 16) >> 6;
  const int tk0 = (mt * 16 + hi8) & 63;
#pragma unroll
  for (int t = 0; t < 4; ++t) {
    const int c     = (ng * 4 + t) * 16 + col;
    const int which = c / kC;
    const int rem   = c - which * kC;
    const int head  = rem >> 5;
    const int d     = rem & 31;
    const float bs  = bias[c];
    const size_t base = (((size_t)win * 3 + which) * kNH + head) * (kN * kHD);
    if (which == 2) {           // V: transposed [d][tok], packed 16B store
      v8h pk;
#pragma unroll
      for (int j = 0; j < 8; ++j) pk[j] = (_Float16)(acc[t][j] + bs);
      *(v8h*)(qkv + base + (size_t)d * kN + tk0) = pk;
    } else {
      const float sc = (which == 0) ? kSCALE : 1.f;
#pragma unroll
      for (int j = 0; j < 8; ++j)
        qkv[base + (size_t)(tk0 + j) * kHD + d] = (_Float16)((acc[t][j] + bs) * sc);
    }
  }
}

// ---------------------------------------------------------------------------
// Windowed attention: one wave per (window, head). 32 WMMAs / wave.
// K and V^T staged in LDS via GLOBAL_LOAD_ASYNC_TO_LDS_B128 (4x reuse each).
// ---------------------------------------------------------------------------
#define SS_LD 65
#define SP_LD 72
__global__ __launch_bounds__(32) void attn_kernel(const _Float16* __restrict__ qkv,
                                                  const float* __restrict__ rpb,
                                                  _Float16* __restrict__ out) {
  __shared__ float    sS[64 * SS_LD];
  __shared__ _Float16 sP[64 * SP_LD];
  __shared__ _Float16 sK[64 * kHD];   // [tok][d]
  __shared__ _Float16 sV[kHD * kN];   // [d][tok]
  __shared__ int      sId[64];
  const int lane = threadIdx.x;
  const int g    = blockIdx.x;
  const int head = blockIdx.y;
  const int wi = g & 255;
  const int hb = wi >> 4, wb = wi & 15;
  const _Float16* Q  = qkv + ((((size_t)g * 3 + 0) * kNH + head) << 11);
  const _Float16* K  = qkv + ((((size_t)g * 3 + 1) * kNH + head) << 11);
  const _Float16* Vt = qkv + ((((size_t)g * 3 + 2) * kNH + head) << 11);

  // kick off async K/V^T staging (ASYNCcnt), overlap with mask-id setup
  async_copy_wave(K, sK, kN * kHD, lane);
  async_copy_wave(Vt, sV, kHD * kN, lane);

  for (int t = lane; t < 64; t += 32) {
    const int hh = hb * kWS + (t >> 3);
    const int ww = wb * kWS + (t & 7);
    const int rh = (hh < kH - kWS) ? 0 : ((hh < kH - kSH) ? 1 : 2);
    const int rw = (ww < kW - kWS) ? 0 : ((ww < kW - kSH) ? 1 : 2);
    sId[t] = rh * 3 + rw;
  }
  async_wait0();
  __syncthreads();

  const int col = lane & 15;
  const int hi8 = (lane >> 4) * 8;

  // S = Q K^T (+ rel-pos bias + shift mask); K rows B-friendly ([tok][d])
  for (int mt = 0; mt < 4; ++mt) {
    v16h aq = load_frag_a(Q + mt * 16 * kHD, kHD, lane);
    for (int nt = 0; nt < 4; ++nt) {
      v16h bk = load_frag_b_rows(sK + (size_t)(nt * 16) * kHD, kHD, lane);
      v8f s = {};
      s = WMMA_F16(aq, bk, s);
#pragma unroll
      for (int j = 0; j < 8; ++j) {
        const int r  = mt * 16 + j + hi8;
        const int cc = nt * 16 + col;
        const int idx = ((r >> 3) - (cc >> 3) + 7) * 15 + ((r & 7) - (cc & 7) + 7);
        const float bias = rpb[idx * kNH + head];
        const float msk  = (sId[r] != sId[cc]) ? -100.f : 0.f;
        sS[r * SS_LD + cc] = s[j] + bias + msk;
      }
    }
  }
  __syncthreads();

  // softmax (2 rows per lane), write f16 P
  for (int r = lane; r < 64; r += 32) {
    float mx = -1e30f;
    for (int cc = 0; cc < 64; ++cc) mx = fmaxf(mx, sS[r * SS_LD + cc]);
    float sum = 0.f;
    for (int cc = 0; cc < 64; ++cc) {
      const float e = __expf(sS[r * SS_LD + cc] - mx);
      sS[r * SS_LD + cc] = e;
      sum += e;
    }
    const float inv = 1.f / sum;
    for (int cc = 0; cc < 64; ++cc)
      sP[r * SP_LD + cc] = (_Float16)(sS[r * SS_LD + cc] * inv);
  }
  __syncthreads();

  // O = P @ V, V^T rows B-friendly ([d][tok])
  for (int mt = 0; mt < 4; ++mt) {
    for (int nt = 0; nt < 2; ++nt) {
      v8f o = {};
#pragma unroll
      for (int kc = 0; kc < 2; ++kc) {
        v16h ap = load_frag_a(sP + mt * 16 * SP_LD + kc * 32, SP_LD, lane);
        v16h bv = load_frag_b_rows(sV + (size_t)(nt * 16) * kN + kc * 32, kN, lane);
        o = WMMA_F16(ap, bv, o);
      }
#pragma unroll
      for (int j = 0; j < 8; ++j) {
        const int tok = mt * 16 + j + hi8;
        out[((size_t)g * kN + tok) * kC + head * kHD + nt * 16 + col] = (_Float16)o[j];
      }
    }
  }
}

// ---------------------------------------------------------------------------
// Proj GEMM + window reverse + un-shift + residual -> fp32 x2
// ---------------------------------------------------------------------------
__global__ void proj_gemm_kernel(const _Float16* __restrict__ ao,
                                 const _Float16* __restrict__ wpk,
                                 const float* __restrict__ bias,
                                 const float* __restrict__ xin,
                                 float* __restrict__ x2) {
  const int lane = threadIdx.x & 31;
  const int wv   = threadIdx.x >> 5;
  const int mt   = blockIdx.x * 4 + wv;
  const int ng   = blockIdx.y;   // 0..2
  const _Float16* A = ao + (size_t)mt * 16 * kC;
  constexpr int K32 = kC / 32;
  v8f acc[4] = {};
#pragma unroll
  for (int kc = 0; kc < K32; ++kc) {
    v16h a = load_frag_a(A + kc * 32, kC, lane);
    if (kc + 1 < K32)
      __builtin_prefetch(wpk + ((size_t)(ng * 4) * K32 + kc + 1) * 512, 0, 3);
#pragma unroll
    for (int t = 0; t < 4; ++t) {
      v16h b = load_frag_b_packed(wpk + ((size_t)(ng * 4 + t) * K32 + kc) * 512, lane);
      acc[t] = WMMA_F16(a, b, acc[t]);
    }
  }
  const int col = lane & 15;
  const int hi8 = (lane >> 4) * 8;
#pragma unroll
  for (int t = 0; t < 4; ++t) {
    const int c    = (ng * 4 + t) * 16 + col;
    const float bs = bias[c];
#pragma unroll
    for (int j = 0; j < 8; ++j) {
      const int mrow = mt * 16 + j + hi8;
      const int win = mrow >> 6, tok = mrow & 63;
      const int img = win >> 8, wwi = win & 255;
      const int hs = ((wwi >> 4) * kWS + (tok >> 3) + kSH) & (kH - 1);
      const int ws = ((wwi & 15) * kWS + (tok & 7) + kSH) & (kW - 1);
      const size_t idx = ((size_t)img * (kH * kW) + hs * kW + ws) * kC + c;
      x2[idx] = xin[idx] + acc[t][j] + bs;
    }
  }
}

// ---------------------------------------------------------------------------
// LN2 -> f16
// ---------------------------------------------------------------------------
__global__ void ln2_kernel(const float* __restrict__ x2,
                           const float* __restrict__ lw,
                           const float* __restrict__ lb,
                           _Float16* __restrict__ m16) {
  const int lane = threadIdx.x & 31;
  const int wv   = threadIdx.x >> 5;
  const int g    = blockIdx.x * (blockDim.x >> 5) + wv;
  const float* src = x2 + (size_t)g * kC;
  float v[6], s = 0.f, ss = 0.f;
#pragma unroll
  for (int i = 0; i < 6; ++i) {
    v[i] = src[lane + 32 * i];
    s += v[i];
    ss += v[i] * v[i];
  }
#pragma unroll
  for (int off = 16; off; off >>= 1) {
    s  += __shfl_xor(s, off, 32);
    ss += __shfl_xor(ss, off, 32);
  }
  const float mu  = s * (1.f / kC);
  const float inv = rsqrtf(ss * (1.f / kC) - mu * mu + kEPS);
  _Float16* dst = m16 + (size_t)g * kC;
#pragma unroll
  for (int i = 0; i < 6; ++i) {
    const int c = lane + 32 * i;
    dst[c] = (_Float16)((v[i] - mu) * inv * lw[c] + lb[c]);
  }
}

// ---------------------------------------------------------------------------
// FC1: [131072,192] x packed[192,768] + bias + exact GELU -> f16
// ---------------------------------------------------------------------------
__global__ void fc1_gemm_kernel(const _Float16* __restrict__ m16,
                                const _Float16* __restrict__ wpk,
                                const float* __restrict__ bias,
                                _Float16* __restrict__ h16) {
  const int lane = threadIdx.x & 31;
  const int wv   = threadIdx.x >> 5;
  const int mt   = blockIdx.x * 4 + wv;
  const int ng   = blockIdx.y;   // 0..11
  const _Float16* A = m16 + (size_t)mt * 16 * kC;
  constexpr int K32 = kC / 32;
  v8f acc[4] = {};
#pragma unroll
  for (int kc = 0; kc < K32; ++kc) {
    v16h a = load_frag_a(A + kc * 32, kC, lane);
    if (kc + 1 < K32)
      __builtin_prefetch(wpk + ((size_t)(ng * 4) * K32 + kc + 1) * 512, 0, 3);
#pragma unroll
    for (int t = 0; t < 4; ++t) {
      v16h b = load_frag_b_packed(wpk + ((size_t)(ng * 4 + t) * K32 + kc) * 512, lane);
      acc[t] = WMMA_F16(a, b, acc[t]);
    }
  }
  const int col = lane & 15;
  const int hi8 = (lane >> 4) * 8;
#pragma unroll
  for (int t = 0; t < 4; ++t) {
    const int c    = (ng * 4 + t) * 16 + col;
    const float bs = bias[c];
#pragma unroll
    for (int j = 0; j < 8; ++j) {
      const int mrow = mt * 16 + j + hi8;
      float v = acc[t][j] + bs;
      v = 0.5f * v * (1.f + erff(v * 0.70710678118654752f));
      h16[(size_t)mrow * (4 * kC) + c] = (_Float16)v;
    }
  }
}

// ---------------------------------------------------------------------------
// FC2: [131072,768] x packed[768,192] + bias + residual(x2) -> fp32 d_out
// ---------------------------------------------------------------------------
__global__ void fc2_gemm_kernel(const _Float16* __restrict__ h16,
                                const _Float16* __restrict__ wpk,
                                const float* __restrict__ bias,
                                const float* __restrict__ x2,
                                float* __restrict__ out) {
  const int lane = threadIdx.x & 31;
  const int wv   = threadIdx.x >> 5;
  const int mt   = blockIdx.x * 4 + wv;
  const int ng   = blockIdx.y;   // 0..2
  const _Float16* A = h16 + (size_t)mt * 16 * (4 * kC);
  constexpr int K32 = (4 * kC) / 32;   // 24
  v8f acc[4] = {};
#pragma unroll 4
  for (int kc = 0; kc < K32; ++kc) {
    v16h a = load_frag_a(A + kc * 32, 4 * kC, lane);
    if (kc + 1 < K32)
      __builtin_prefetch(wpk + ((size_t)(ng * 4) * K32 + kc + 1) * 512, 0, 3);
#pragma unroll
    for (int t = 0; t < 4; ++t) {
      v16h b = load_frag_b_packed(wpk + ((size_t)(ng * 4 + t) * K32 + kc) * 512, lane);
      acc[t] = WMMA_F16(a, b, acc[t]);
    }
  }
  const int col = lane & 15;
  const int hi8 = (lane >> 4) * 8;
#pragma unroll
  for (int t = 0; t < 4; ++t) {
    const int c    = (ng * 4 + t) * 16 + col;
    const float bs = bias[c];
#pragma unroll
    for (int j = 0; j < 8; ++j) {
      const int mrow = mt * 16 + j + hi8;
      const size_t idx = (size_t)mrow * kC + c;
      out[idx] = x2[idx] + acc[t][j] + bs;
    }
  }
}

// ---------------------------------------------------------------------------
extern "C" void kernel_launch(void* const* d_in, const int* in_sizes, int n_in,
                              void* d_out, int out_size, void* d_ws, size_t ws_size,
                              hipStream_t stream) {
  (void)in_sizes; (void)n_in; (void)out_size; (void)ws_size;
  const float* x      = (const float*)d_in[0];
  const float* ln1_w  = (const float*)d_in[1];
  const float* ln1_b  = (const float*)d_in[2];
  const float* qkv_w  = (const float*)d_in[3];
  const float* qkv_b  = (const float*)d_in[4];
  const float* rpb    = (const float*)d_in[5];
  const float* proj_w = (const float*)d_in[6];
  const float* proj_b = (const float*)d_in[7];
  const float* ln2_w  = (const float*)d_in[8];
  const float* ln2_b  = (const float*)d_in[9];
  const float* fc1_w  = (const float*)d_in[10];
  const float* fc1_b  = (const float*)d_in[11];
  const float* fc2_w  = (const float*)d_in[12];
  const float* fc2_b  = (const float*)d_in[13];
  float* out = (float*)d_out;

  char* ws = (char*)d_ws;
  size_t off = 0;
  auto take = [&](size_t bytes) {
    void* p = ws + off;
    off += (bytes + 255) & ~(size_t)255;
    return p;
  };
  _Float16* qkvw16 = (_Float16*)take((size_t)kC * 3 * kC * 2);
  _Float16* prjw16 = (_Float16*)take((size_t)kC * kC * 2);
  _Float16* fc1w16 = (_Float16*)take((size_t)kC * 4 * kC * 2);
  _Float16* fc2w16 = (_Float16*)take((size_t)4 * kC * kC * 2);
  _Float16* R1     = (_Float16*)take((size_t)kMTOT * kC * 2);      // xw / attnO / m16
  float*    x2     = (float*)take((size_t)kMTOT * kC * 4);         // residual stream
  _Float16* R3     = (_Float16*)take((size_t)kMTOT * 4 * kC * 2);  // qkv / h16

  _Float16* xw  = R1;
  _Float16* qkv = R3;
  _Float16* ao  = R1;
  _Float16* m16 = R1;
  _Float16* h16 = R3;

  // 1) weight conversion + B-fragment packing
  pack_b_kernel<<<(kC * 3 * kC + 255) / 256, 256, 0, stream>>>(qkv_w, qkvw16, kC, 3 * kC);
  pack_b_kernel<<<(kC * kC + 255) / 256, 256, 0, stream>>>(proj_w, prjw16, kC, kC);
  pack_b_kernel<<<(kC * 4 * kC + 255) / 256, 256, 0, stream>>>(fc1_w, fc1w16, kC, 4 * kC);
  pack_b_kernel<<<(4 * kC * kC + 255) / 256, 256, 0, stream>>>(fc2_w, fc2w16, 4 * kC, kC);

  // 2) LN1 + shift + partition
  ln1_shift_part_kernel<<<kMTOT / 8, 256, 0, stream>>>(x, ln1_w, ln1_b, xw);

  // 3) QKV GEMM
  qkv_gemm_kernel<<<dim3(2048, 9), 128, 0, stream>>>(xw, qkvw16, qkv_b, qkv);

  // 4) attention
  attn_kernel<<<dim3(kWTOT, kNH), 32, 0, stream>>>(qkv, rpb, ao);

  // 5) proj + reverse/unshift + residual
  proj_gemm_kernel<<<dim3(2048, 3), 128, 0, stream>>>(ao, prjw16, proj_b, x, x2);

  // 6) LN2
  ln2_kernel<<<kMTOT / 8, 256, 0, stream>>>(x2, ln2_w, ln2_b, m16);

  // 7) FC1 + GELU
  fc1_gemm_kernel<<<dim3(2048, 12), 128, 0, stream>>>(m16, fc1w16, fc1_b, h16);

  // 8) FC2 + residual -> d_out
  fc2_gemm_kernel<<<dim3(2048, 3), 128, 0, stream>>>(h16, fc2w16, fc2_b, x2, out);
}
